// FeatureWarperSoftsplat_62526133895473
// MI455X (gfx1250) — matrix operand order repaired and checked by reading, not compiled, not run on hardware
//
#include <hip/hip_runtime.h>
#include <math.h>

// ---- CDNA5 vector types ----
typedef _Float16 v16h __attribute__((ext_vector_type(16)));
typedef _Float16 v8h  __attribute__((ext_vector_type(8)));
typedef float    v8f  __attribute__((ext_vector_type(8)));
typedef unsigned int v4u __attribute__((ext_vector_type(4)));
typedef int v8i __attribute__((ext_vector_type(8)));
typedef int v4i __attribute__((ext_vector_type(4)));

#define BATCH 16
#define CIN   128
#define OC    64
#define HH    128
#define WW    128
#define HWSZ  (HH*WW)

// -----------------------------------------------------------------------------
// TDM descriptor builder + issue (2D tile, optional LDS padding).
// D# group layout per CDNA5 ISA ch.8 (CDNA5 ISA).
//   dsz: data_size code (0=1B,1=2B,2=4B,3=8B); dims/strides in data_size units.
// -----------------------------------------------------------------------------
__device__ __forceinline__ void tdm_load_2d(unsigned lds_off, const void* gptr,
                                            unsigned dsz, unsigned td0, unsigned td1,
                                            unsigned tl0, unsigned tl1,
                                            unsigned long long stride0,
                                            unsigned pad_en, unsigned pad_int,
                                            unsigned pad_amt) {
    unsigned long long ga = (unsigned long long)gptr;
    v4u g0;
    g0.x = 1u;                                   // count=1, user mode
    g0.y = lds_off;                              // lds_addr (bytes)
    g0.z = (unsigned)ga;                         // global_addr[31:0]
    g0.w = (unsigned)(ga >> 32) | (2u << 30);    // global_addr[56:32] | type=2
    v8i g1;
    g1[0] = (int)((dsz << 16) | (pad_en << 20) | (pad_int << 22) | (pad_amt << 25));
    g1[1] = (int)((td0 & 0xffffu) << 16);                         // tensor_dim0 lo
    g1[2] = (int)((td0 >> 16) | ((td1 & 0xffffu) << 16));         // dim0 hi | dim1 lo
    g1[3] = (int)((td1 >> 16) | (tl0 << 16));                     // dim1 hi | tile_dim0
    g1[4] = (int)(tl1 & 0xffffu);                                 // tile_dim1 (tile_dim2=0)
    g1[5] = (int)(unsigned)(stride0 & 0xffffffffull);             // tensor_dim0_stride lo
    g1[6] = (int)(unsigned)((stride0 >> 32) & 0xffffu);           // stride hi (dim1_stride=0)
    g1[7] = 0;
    v4i g2 = (v4i)0, g3 = (v4i)0;
#if defined(__clang_major__) && __clang_major__ >= 23
    v8i g4 = (v8i)0;
    __builtin_amdgcn_tensor_load_to_lds(g0, g1, g2, g3, g4, 0);
#else
    __builtin_amdgcn_tensor_load_to_lds(g0, g1, g2, g3, 0);
#endif
}

// -----------------------------------------------------------------------------
// Kernel 1: repack w1 (f32 [oc][c][ky][kx]) into f16 WMMA B-fragments.
// wB[p][kk][t][lane][e]: k = kk*32 + (lane<16 ? e : 16+e), n = lane%16,
// oc = t*16+n  ->  one contiguous 32B load per lane per fragment.
// -----------------------------------------------------------------------------
__global__ void prep_w1_kernel(const float* __restrict__ w1, _Float16* __restrict__ wB) {
    int i = blockIdx.x * 256 + threadIdx.x;          // 9*4*4*32*16 = 73728
    if (i >= 9*4*4*32*16) return;
    int e    = i & 15;
    int lane = (i >> 4) & 31;
    int t    = (i >> 9) & 3;
    int kk   = (i >> 11) & 3;
    int p    = i >> 13;
    int ky = p / 3, kx = p % 3;
    int k  = kk * 32 + ((lane >> 4) ? (16 + e) : e);
    int oc = t * 16 + (lane & 15);
    wB[i] = (_Float16)w1[((oc * CIN + k) * 3 + ky) * 3 + kx];
}

// -----------------------------------------------------------------------------
// Kernel 2: feat f32 NCHW -> f16 pixel-major [b][y][x][c] (LDS transpose so both
// global read and write are coalesced 16B). Enables TDM staging in conv1.
// -----------------------------------------------------------------------------
__global__ __launch_bounds__(256) void transpose_feat_kernel(
        const float* __restrict__ feat, _Float16* __restrict__ featT) {
    __shared__ _Float16 sT[128][136];                // pad 136 halves = 272B (16B mult.)
    const int by = blockIdx.x, b = by >> 7, y = by & 127;
    const int tid = threadIdx.x;
    for (int u = tid; u < 128 * 32; u += 256) {      // c x quad-of-x
        int c = u >> 5, x0 = (u & 31) << 2;
        const float* src = feat + (((size_t)b * CIN + c) * HH + y) * WW + x0;
        float4 v = *(const float4*)src;
        if (y + 1 < HH) __builtin_prefetch(src + WW, 0, 0);   // global_prefetch_b8
        sT[x0 + 0][c] = (_Float16)v.x;
        sT[x0 + 1][c] = (_Float16)v.y;
        sT[x0 + 2][c] = (_Float16)v.z;
        sT[x0 + 3][c] = (_Float16)v.w;
    }
    __syncthreads();
    for (int u = tid; u < 128 * 16; u += 256) {      // x x 8-channel segment
        int x = u >> 4, seg = u & 15;
        *(v8h*)(featT + (((size_t)b * HH + y) * WW + x) * CIN + seg * 8) =
            *(const v8h*)&sT[x][seg * 8];
    }
}

// -----------------------------------------------------------------------------
// Kernel 3: conv1 (3x3, 128->64) + bias + ReLU, implicit GEMM on WMMA f16.
// Block = 4 waves = one (b,y) output row; wave -> 32 pixels (2 M-tiles) x 64 oc
// (4 N-tiles). K = 4 channel chunks of 32 x 9 taps.
// Staging is ALL Tensor-Data-Mover: input tile (with TDM LDS padding for bank-
// conflict-free A loads) + per-tap double-buffered weight fragments, pipelined
// one tap ahead on TENSORcnt.
// -----------------------------------------------------------------------------
__global__ __launch_bounds__(128) void conv1_wmma_kernel(
        const _Float16* __restrict__ featT, const _Float16* __restrict__ wB,
        const float* __restrict__ b1, _Float16* __restrict__ h) {
    // [row 0..2][col 0..129 (x=-1..128)][c 0..31 pad->40 halves]
    // 40 halves = 80B = 20 dwords/col -> 16 distinct bank starts spaced 4:
    // conflict-free b128 A-fragment loads. TDM pad: 64B data + 16B pad per line.
    __shared__ __attribute__((aligned(16))) _Float16 sIn[3][130][40];
    __shared__ __attribute__((aligned(32))) _Float16 sW[2][2048];  // tap double-buffer

    const int by = blockIdx.x, b = by >> 7, y = by & 127;
    const int tid = threadIdx.x, wave = tid >> 5, lane = tid & 31;
    const int l16 = lane & 15, hi = lane >> 4;

    v8f acc[2][4];
    #pragma unroll
    for (int mt = 0; mt < 2; ++mt)
        #pragma unroll
        for (int t = 0; t < 4; ++t) acc[mt][t] = (v8f)0.0f;

    for (int cc = 0; cc < 4; ++cc) {                 // 32-channel K chunks
        __syncthreads();                             // prior chunk done with sIn/sW
        if (wave == 0) {
            // input rows y-1..y+1 via TDM: 2B units, line = 32 ch, 128 lines (x),
            // row stride 128 units; pad 16B after every 64B -> LDS stride 40 halves
            for (int r = 0; r < 3; ++r) {
                int gy = y + r - 1;
                if (gy >= 0 && gy < HH)
                    tdm_load_2d((unsigned)(size_t)&sIn[r][1][0],
                                featT + (((size_t)b * HH + gy) * WW) * CIN + cc * 32,
                                /*dsz*/1, /*td0*/32, /*td1*/128, /*tl0*/32, /*tl1*/128,
                                /*stride0*/128, /*pad*/1, /*int:64B*/3, /*amt:16B*/3);
            }
            // tap-0 weight fragments (4 KB contiguous, 8B units)
            tdm_load_2d((unsigned)(size_t)&sW[0][0], wB + (size_t)(0 * 4 + cc) * 2048,
                        3, 512, 1, 512, 1, 512, 0, 0, 0);
            __builtin_amdgcn_s_wait_tensorcnt(0);
        }
        // zero out-of-image row and halo columns (disjoint from TDM writes)
        {
            int r_oob = (y == 0) ? 0 : ((y == 127) ? 2 : -1);
            if (r_oob >= 0)
                for (int u = tid; u < 130 * 5; u += 128)
                    *(v8h*)&sIn[r_oob][u / 5][(u % 5) * 8] = (v8h)(_Float16)0.f;
            for (int u = tid; u < 30; u += 128) {    // cols x=-1 and x=128
                int r = u / 10, side = (u / 5) & 1, s = u % 5;
                *(v8h*)&sIn[r][side ? 129 : 0][s * 8] = (v8h)(_Float16)0.f;
            }
        }
        __syncthreads();                             // sIn + sW[0] published

        for (int p = 0; p < 9; ++p) {
            if (p < 8 && wave == 0)                  // prefetch next tap's weights
                tdm_load_2d((unsigned)(size_t)&sW[(p + 1) & 1][0],
                            wB + (size_t)((p + 1) * 4 + cc) * 2048,
                            3, 512, 1, 512, 1, 512, 0, 0, 0);
            const int ky = p / 3, kx = p % 3;

            v16h afrag[2];
            #pragma unroll
            for (int mt = 0; mt < 2; ++mt) {         // A 16x32 f16 layout
                const int x   = wave * 32 + mt * 16 + l16;    // M = lane%16
                const int col = x + kx;                       // halo-shifted column
                const int cb  = hi ? 16 : 0;
                v8h lo = *(const v8h*)&sIn[ky][col][cb];
                v8h hh = *(const v8h*)&sIn[ky][col][cb + 8];
                afrag[mt] = __builtin_shufflevector(lo, hh,
                    0,1,2,3,4,5,6,7,8,9,10,11,12,13,14,15);
            }
            const _Float16* sb = &sW[p & 1][0];
            #pragma unroll
            for (int t = 0; t < 4; ++t) {
                v16h bfrag = *(const v16h*)&sb[(t * 32 + lane) * 16];
                acc[0][t] = __builtin_amdgcn_wmma_f32_16x16x32_f16(
                    false, afrag[0], false, bfrag, (short)0, acc[0][t], false, false);
                acc[1][t] = __builtin_amdgcn_wmma_f32_16x16x32_f16(
                    false, afrag[1], false, bfrag, (short)0, acc[1][t], false, false);
            }
            if (p < 8) {
                if (wave == 0) __builtin_amdgcn_s_wait_tensorcnt(0);
                __syncthreads();                     // next tap's weights visible
            }
        }
    }

    // epilogue: bias + ReLU, h f16 pixel-major [b][y][x][oc]
    // C/D layout: N = lane%16, M = e + 8*(lane>=16)
    #pragma unroll
    for (int mt = 0; mt < 2; ++mt)
        #pragma unroll
        for (int t = 0; t < 4; ++t) {
            const int oc = t * 16 + l16;
            const float bias = b1[oc];
            #pragma unroll
            for (int e = 0; e < 8; ++e) {
                const int x = wave * 32 + mt * 16 + e + (hi ? 8 : 0);
                float v = acc[mt][t][e] + bias;
                v = v > 0.f ? v : 0.f;
                h[((size_t)by * WW + x) * OC + oc] = (_Float16)v;
            }
        }
}

// -----------------------------------------------------------------------------
// Kernel 4: conv2 (3x3, 64->1) + bias + exp. Staging via async global->LDS DMA
// (h is pixel-major f16, so each 16B chunk maps 1:1), completed on ASYNCcnt.
// -----------------------------------------------------------------------------
__global__ __launch_bounds__(128) void conv2_exp_kernel(
        const _Float16* __restrict__ h, const float* __restrict__ w2,
        const float* __restrict__ b2, float* __restrict__ expm) {
    __shared__ __attribute__((aligned(16))) _Float16 sH[3][130][72];
    const int by = blockIdx.x, b = by >> 7, y = by & 127;
    const int tid = threadIdx.x;

    for (int u = tid; u < 3 * 128 * 8; u += 128) {
        int r = u >> 10, x = (u >> 3) & 127, seg = u & 7;
        int gy = y + r - 1;
        if (gy >= 0 && gy < HH) {
            unsigned lds = (unsigned)(size_t)&sH[r][x + 1][seg * 8];
            unsigned long long ga = (unsigned long long)
                (h + (((size_t)b * HH + gy) * WW + x) * OC + seg * 8);
            asm volatile("global_load_async_to_lds_b128 %0, %1, off"
                         :: "v"(lds), "v"(ga) : "memory");
        } else {
            *(v8h*)&sH[r][x + 1][seg * 8] = (v8h)(_Float16)0.f;
        }
    }
    for (int u = tid; u < 3 * 2 * 8; u += 128) {     // halo columns
        int r = u >> 4, side = (u >> 3) & 1, seg = u & 7;
        *(v8h*)&sH[r][side ? 129 : 0][seg * 8] = (v8h)(_Float16)0.f;
    }
    asm volatile("s_wait_asynccnt 0x0" ::: "memory");
    __syncthreads();

    const int x = tid;
    float m = b2[0];
    for (int p = 0; p < 9; ++p) {
        const int ky = p / 3, kx = p % 3;
        const _Float16* row = &sH[ky][x + kx][0];
        #pragma unroll 8
        for (int oc = 0; oc < OC; ++oc)
            m += (float)row[oc] * w2[(oc * 3 + ky) * 3 + kx];   // uniform -> s_load
    }
    expm[(size_t)by * WW + x] = expf(m);
}

// -----------------------------------------------------------------------------
// Kernel 5: softsplat scatter (bilinear forward warp with atomics).
// -----------------------------------------------------------------------------
__global__ __launch_bounds__(256) void scatter_kernel(
        const float* __restrict__ feat, const float* __restrict__ flow,
        const float* __restrict__ expm, float* __restrict__ out,
        float* __restrict__ norm) {
    const int g  = blockIdx.x * 256 + threadIdx.x;   // B*HW = 262144
    const int b  = g >> 14;
    const int hw = g & (HWSZ - 1);
    const int y  = hw >> 7;
    const int x  = hw & 127;

    const float fx = (float)x + flow[((size_t)b * 2 + 0) * HWSZ + hw];
    const float fy = (float)y + flow[((size_t)b * 2 + 1) * HWSZ + hw];
    const float x0f = floorf(fx), y0f = floorf(fy);
    const int   x0  = (int)x0f,  y0  = (int)y0f;
    const float wx1 = fx - x0f,  wy1 = fy - y0f;
    const float wx0 = 1.f - wx1, wy0 = 1.f - wy1;

    int idx[4]; float w[4]; bool ok[4];
    const int   xs[2] = { x0, x0 + 1 }, ys[2] = { y0, y0 + 1 };
    const float wxs[2] = { wx0, wx1 },  wys[2] = { wy0, wy1 };
    #pragma unroll
    for (int j = 0; j < 4; ++j) {
        const int xi = xs[j & 1], yi = ys[j >> 1];
        ok[j]  = (xi >= 0) & (xi < WW) & (yi >= 0) & (yi < HH);
        idx[j] = yi * WW + xi;
        w[j]   = wxs[j & 1] * wys[j >> 1];
    }

    const float e = expm[g];
    #pragma unroll
    for (int j = 0; j < 4; ++j)
        if (ok[j]) atomicAdd(&norm[(size_t)b * HWSZ + idx[j]], w[j] * e);

    for (int c = 0; c < CIN; ++c) {
        const float val = feat[((size_t)b * CIN + c) * HWSZ + hw] * e;
        float* outc = out + ((size_t)b * CIN + c) * HWSZ;
        #pragma unroll
        for (int j = 0; j < 4; ++j)
            if (ok[j]) atomicAdd(&outc[idx[j]], val * w[j]);
    }
}

// -----------------------------------------------------------------------------
// Kernel 6: in-place normalize d_out by norm plane (0 -> 1).
// -----------------------------------------------------------------------------
__global__ __launch_bounds__(256) void normalize_kernel(
        float* __restrict__ out, const float* __restrict__ norm) {
    const int i  = blockIdx.x * 256 + threadIdx.x;   // 33,554,432
    const int b  = i >> 21;
    const int hw = i & (HWSZ - 1);
    const float n = norm[(size_t)b * HWSZ + hw];
    out[i] = out[i] / (n == 0.f ? 1.f : n);
}

__global__ void fill_zero_kernel(float* __restrict__ p, int n) {
    int i = blockIdx.x * 256 + threadIdx.x;
    if (i < n) p[i] = 0.f;
}

// -----------------------------------------------------------------------------
extern "C" void kernel_launch(void* const* d_in, const int* in_sizes, int n_in,
                              void* d_out, int out_size, void* d_ws, size_t ws_size,
                              hipStream_t stream) {
    const float* feat = (const float*)d_in[0];
    const float* flow = (const float*)d_in[1];
    const float* w1   = (const float*)d_in[2];
    const float* b1   = (const float*)d_in[3];
    const float* w2   = (const float*)d_in[4];
    const float* b2   = (const float*)d_in[5];
    float* out = (float*)d_out;

    char* ws = (char*)d_ws;
    const size_t WB_BYTES    = (size_t)9 * 4 * 4 * 32 * 16 * sizeof(_Float16); // 147,456
    const size_t FEATT_BYTES = (size_t)BATCH * HWSZ * CIN * sizeof(_Float16);  // 67.1 MB
    const size_t H_BYTES     = (size_t)BATCH * OC * HWSZ * sizeof(_Float16);   // 33.5 MB
    const size_t EXPM_BYTES  = (size_t)BATCH * HWSZ * sizeof(float);           // 1 MB
    _Float16* wB    = (_Float16*)(ws);
    _Float16* featT = (_Float16*)(ws + WB_BYTES);
    _Float16* hbuf  = (_Float16*)(ws + WB_BYTES + FEATT_BYTES);
    float*    expm  = (float*)   (ws + WB_BYTES + FEATT_BYTES + H_BYTES);
    float*    normp = (float*)   (ws + WB_BYTES + FEATT_BYTES + H_BYTES + EXPM_BYTES);

    const int nOut  = BATCH * CIN * HWSZ;   // 33,554,432
    const int nNorm = BATCH * HWSZ;         // 262,144

    prep_w1_kernel<<<(9*4*4*32*16 + 255) / 256, 256, 0, stream>>>(w1, wB);
    transpose_feat_kernel<<<BATCH * HH, 256, 0, stream>>>(feat, featT);
    fill_zero_kernel<<<(nOut  + 255) / 256, 256, 0, stream>>>(out,   nOut);
    fill_zero_kernel<<<(nNorm + 255) / 256, 256, 0, stream>>>(normp, nNorm);

    conv1_wmma_kernel<<<BATCH * HH, 128, 0, stream>>>(featT, wB, b1, hbuf);
    conv2_exp_kernel <<<BATCH * HH, 128, 0, stream>>>(hbuf, w2, b2, expm);
    scatter_kernel   <<<nNorm / 256, 256, 0, stream>>>(feat, flow, expm, out, normp);
    normalize_kernel <<<nOut  / 256, 256, 0, stream>>>(out, normp);
}
